// UnimolCoordHead_47218870453083
// MI455X (gfx1250) — compile-verified, matrix operand order checked by reading, not run on hardware
//
#include <hip/hip_runtime.h>
#include <hip/hip_fp16.h>

// UnimolCoordHead fused kernel for gfx1250 (MI455X).
// B=4, L=512, D=128. x streamed once (536 MB, ~23 us @ 23.3 TB/s); GEMM
// (34.4 GFLOP) on v_wmma_f32_16x16x32_f16; GELU on 134M hidden elements is the
// VALU hot spot -> branch-free A&S erf (|err|<=1.5e-7, below f16-GEMM noise)
// using raw v_rcp_f32 / v_exp_f32. Split dscnt-only barriers keep the next
// x-tile's global loads in flight across the whole WMMA/GELU phase.

typedef __attribute__((ext_vector_type(16))) _Float16 v16h;
typedef __attribute__((ext_vector_type(8)))  float    v8f;

#define BATCH 4
#define LSEQ  512
#define DDIM  128
#define TILE  16
#define NTILES (LSEQ / TILE)   // 32 j-tiles per (b,i)

// 0.5*v*(1+erf(v/sqrt(2))) with Abramowitz-Stegun 7.1.26 erf (max err 1.5e-7).
// Branch-free: ~14 VALU + v_rcp_f32 + v_exp_f32, no exec-mask divergence.
__device__ __forceinline__ float gelu_erf_fast(float v) {
    const float x  = v * 0.70710678118654752f;
    const float ax = fabsf(x);
    const float t  = __builtin_amdgcn_rcpf(fmaf(0.3275911f, ax, 1.0f));
    float p = fmaf(1.061405429f, t, -1.453152027f);
    p = fmaf(p, t, 1.421413741f);
    p = fmaf(p, t, -0.284496736f);
    p = fmaf(p, t, 0.254829592f);
    p = p * t;
    const float e      = __expf(-ax * ax);        // v_exp_f32 (exp2 scaled)
    const float erf_ax = fmaf(-p, e, 1.0f);
    const float erf_x  = copysignf(erf_ax, x);
    return 0.5f * v * (1.0f + erf_x);
}

// Workgroup barrier publishing only LDS traffic (s_wait_dscnt 0): outstanding
// global loads (next-tile x stream) stay in flight.
__device__ __forceinline__ void lds_barrier() {
    asm volatile(
        "s_wait_dscnt 0x0\n\t"
        "s_barrier_signal -1\n\t"
        "s_barrier_wait -1"
        ::: "memory");
}

__global__ __launch_bounds__(128)
void unimol_coord_head_kernel(const float* __restrict__ x,
                              const float* __restrict__ coord,
                              const long long* __restrict__ nodes,   // jnp.int64
                              const float* __restrict__ ln_g,
                              const float* __restrict__ ln_b,
                              const float* __restrict__ w1,   // [D,D] (d -> e)
                              const float* __restrict__ b1,   // [D]
                              const float* __restrict__ w2,   // [D,1]
                              const float* __restrict__ b2,   // [1]
                              float* __restrict__ out)        // [B,L,3]
{
    __shared__ __align__(16) _Float16 w1h[DDIM * DDIM];       // 32 KB f16 weights
    __shared__ __align__(16) _Float16 hbuf[2][TILE * DDIM];   // 2 x 4 KB LN'd tiles
    __shared__ float attnpart[2][4 * TILE];                   // per-wave attn partials
    __shared__ float lng[DDIM], lnb[DDIM];
    __shared__ float cjall[LSEQ * 3];                         // coord[b,:,:]  (6 KB)
    __shared__ float npall[LSEQ];                             // not-pad flags (2 KB)
    __shared__ float wred[4];                                 // atom-count partials
    __shared__ float cured[48];                               // final fold

    const int tid  = threadIdx.x;
    const int wave = tid >> 5;
    const int lane = tid & 31;
    const int g    = blockIdx.x;       // one workgroup per (b, i)
    const int b    = g >> 9;
    const int i    = g & (LSEQ - 1);

    // phase-1 lane mapping (8 lanes per row, 16 elements each)
    const int row = tid >> 3;          // 0..15  (j within tile)
    const int s   = tid & 7;

    // ---- LN params to LDS ----
    if (tid < DDIM) { lng[tid] = ln_g[tid]; lnb[tid] = ln_b[tid]; }

    // ---- w1 f32 -> f16 into LDS (w1 is L2-resident: ~free re-read) ----
    for (int e = tid * 4; e < DDIM * DDIM; e += 128 * 4) {
        float4 wv = *(const float4*)(w1 + e);
        union { _Float16 h[2]; unsigned int u; } p0, p1;
        p0.h[0] = (_Float16)wv.x; p0.h[1] = (_Float16)wv.y;
        p1.h[0] = (_Float16)wv.z; p1.h[1] = (_Float16)wv.w;
        *(unsigned int*)(w1h + e)     = p0.u;
        *(unsigned int*)(w1h + e + 2) = p1.u;
    }

    // ---- coord[b,:,:] to LDS ----
    for (int e = tid; e < LSEQ * 3; e += 128) cjall[e] = coord[b * LSEQ * 3 + e];

    // ---- not-pad flags + atom_num[b] (single pass over nodes) ----
    int cnt = 0;
    #pragma unroll
    for (int q = 0; q < 4; ++q) {
        const int idx = tid + q * 128;
        const int np  = (nodes[b * LSEQ + idx] != 0) ? 1 : 0;
        npall[idx] = (float)np;
        cnt += np;
    }
    #pragma unroll
    for (int m = 1; m < 32; m <<= 1) cnt += __shfl_xor(cnt, m);
    if (lane == 0) wred[wave] = (float)cnt;
    __syncthreads();
    const float inv_atom = 1.0f / (wred[0] + wred[1] + wred[2] + wred[3]);
    const float npi = npall[i];
    const float b2v = b2[0];

    // ---- persistent B fragments (w1 slices) for this wave ----
    // wave owns output columns [32*wave, 32*wave+32): two 16-col n-tiles.
    // 16-bit B layout (mirror of A): lane holds column n=lane%16; lane-half
    // selects K subset {0..7,16..23} vs {8..15,24..31} (+32*kk).
    v16h bfrag[2][4];
    {
        const int n_lo = lane & 15;
        const int hh   = lane >> 4;
        #pragma unroll
        for (int nt = 0; nt < 2; ++nt) {
            const int n = wave * 32 + nt * 16 + n_lo;
            #pragma unroll
            for (int kk = 0; kk < 4; ++kk) {
                v16h f;
                #pragma unroll
                for (int t = 0; t < 16; ++t) {
                    const int K = kk * 32 + ((t < 8) ? (hh * 8 + t)
                                                     : (16 + hh * 8 + (t - 8)));
                    f[t] = w1h[K * DDIM + n];
                }
                bfrag[nt][kk] = f;
            }
        }
    }

    const int   n0   = wave * 32 + (lane & 15);
    const float b1v0 = b1[n0],      b1v1 = b1[n0 + 16];
    const float w2v0 = w2[n0],      w2v1 = w2[n0 + 16];

    float ci = 0.0f;
    if (tid < 48) ci = cjall[i * 3 + (tid % 3)];
    float cu = 0.0f;

    const size_t xrowbase = ((size_t)(b * LSEQ + i)) * LSEQ * DDIM;

    // ---- prime the register pipeline with tile 0 ----
    union XReg { float4 q[4]; float f[16]; } xv;
    {
        const float* xr = x + xrowbase + (size_t)row * DDIM + s * 16;
        xv.q[0] = ((const float4*)xr)[0];
        xv.q[1] = ((const float4*)xr)[1];
        xv.q[2] = ((const float4*)xr)[2];
        xv.q[3] = ((const float4*)xr)[3];
    }

    for (int jt = 0; jt < NTILES; ++jt) {
        const int jbase = jt * TILE;
        const int p     = jt & 1;

        // ============ Phase 1: LayerNorm current tile (regs) -> f16 LDS ======
        {
            float sum = 0.0f, ss = 0.0f;
            #pragma unroll
            for (int t = 0; t < 16; ++t) { sum += xv.f[t]; ss += xv.f[t] * xv.f[t]; }
            #pragma unroll
            for (int m = 1; m < 8; m <<= 1) {
                sum += __shfl_xor(sum, m);
                ss  += __shfl_xor(ss,  m);
            }
            const float mu   = sum * (1.0f / 128.0f);
            const float var  = ss * (1.0f / 128.0f) - mu * mu;
            const float rstd = rsqrtf(var + 1e-5f);
            const float np_j = npall[jbase + row];

            _Float16* hb = hbuf[p] + row * DDIM + s * 16;
            const float* gg = lng + s * 16;
            const float* bb = lnb + s * 16;
            #pragma unroll
            for (int t = 0; t < 16; t += 2) {
                const float a0 = ((xv.f[t]     - mu) * rstd * gg[t]     + bb[t]    ) * np_j;
                const float a1 = ((xv.f[t + 1] - mu) * rstd * gg[t + 1] + bb[t + 1]) * np_j;
                union { _Float16 h[2]; unsigned int u; } pk;
                pk.h[0] = (_Float16)a0; pk.h[1] = (_Float16)a1;
                *(unsigned int*)(hb + t) = pk.u;
            }

            // refill x registers for tile jt+1: these global loads remain in
            // flight across both (dscnt-only) barriers and the WMMA phase.
            if (jt + 1 < NTILES) {
                const float* xn = x + xrowbase
                                + (size_t)((jt + 1) * TILE + row) * DDIM + s * 16;
                xv.q[0] = ((const float4*)xn)[0];
                xv.q[1] = ((const float4*)xn)[1];
                xv.q[2] = ((const float4*)xn)[2];
                xv.q[3] = ((const float4*)xn)[3];
                if (jt + 2 < NTILES)   // L2-ahead stream: global_prefetch_b8
                    __builtin_prefetch((const void*)(xn + TILE * DDIM), 0, 1);
            }
        }
        lds_barrier();   // barrier 1: hbuf[p] visible (dscnt only)

        // ========== Phase 2: WMMA GEMM (16 x 128 x 128) + GELU + w2 dot ======
        {
            const int m  = lane & 15;
            const int hh = lane >> 4;

            // load all 4 A fragments as one ds clause, then run the WMMA chain
            union AF { v16h v; uint4 u[2]; } af[4];
            #pragma unroll
            for (int kk = 0; kk < 4; ++kk) {
                af[kk].u[0] = *(const uint4*)(hbuf[p] + m * DDIM + kk * 32 + hh * 8);
                af[kk].u[1] = *(const uint4*)(hbuf[p] + m * DDIM + kk * 32 + 16 + hh * 8);
            }
            v8f acc0 = {}; v8f acc1 = {};
            #pragma unroll
            for (int kk = 0; kk < 4; ++kk) {
                acc0 = __builtin_amdgcn_wmma_f32_16x16x32_f16(
                           false, af[kk].v, false, bfrag[0][kk], (short)0, acc0, false, false);
                acc1 = __builtin_amdgcn_wmma_f32_16x16x32_f16(
                           false, af[kk].v, false, bfrag[1][kk], (short)0, acc1, false, false);
            }
            // C layout: lanes 0-15 hold rows r (VGPR r), lanes 16-31 rows 8+r.
            #pragma unroll
            for (int r = 0; r < 8; ++r) {
                float t = gelu_erf_fast(acc0[r] + b1v0) * w2v0
                        + gelu_erf_fast(acc1[r] + b1v1) * w2v1;
                #pragma unroll
                for (int mm = 1; mm < 16; mm <<= 1) t += __shfl_xor(t, mm);
                if ((lane & 15) == 0)
                    attnpart[p][wave * 16 + (lane >> 4) * 8 + r] = t;
            }
        }
        lds_barrier();   // barrier 2: attnpart[p] visible (dscnt only)

        // ===== Phase 3: fold attn into coord update (runs alongside next ph1) =
        if (tid < 48) {
            const int r = tid / 3;
            const float attn = attnpart[p][r] + attnpart[p][16 + r]
                             + attnpart[p][32 + r] + attnpart[p][48 + r] + b2v;
            cu += (cjall[(jbase + r) * 3 + tid % 3] - ci)
                  * (inv_atom * attn * npi * npall[jbase + r]);
        }
        // no third barrier: next iteration uses parity 1-p buffers
    }

    // ---- final fold ----
    __syncthreads();
    if (tid < 48) cured[tid] = cu;
    __syncthreads();
    if (tid < 3) {
        float ssum = 0.0f;
        #pragma unroll
        for (int r = 0; r < 16; ++r) ssum += cured[r * 3 + tid];
        out[(b * LSEQ + i) * 3 + tid] = ci + ssum;   // ci == coord[b,i,tid] here
    }
}

extern "C" void kernel_launch(void* const* d_in, const int* in_sizes, int n_in,
                              void* d_out, int out_size, void* d_ws, size_t ws_size,
                              hipStream_t stream) {
    (void)in_sizes; (void)n_in; (void)out_size; (void)d_ws; (void)ws_size;
    const float*     x     = (const float*)d_in[0];
    const float*     coord = (const float*)d_in[1];
    const long long* nodes = (const long long*)d_in[2];   // jnp.int64
    const float*     ln_g  = (const float*)d_in[3];
    const float*     ln_b  = (const float*)d_in[4];
    const float*     w1    = (const float*)d_in[5];
    const float*     b1    = (const float*)d_in[6];
    const float*     w2    = (const float*)d_in[7];
    const float*     b2    = (const float*)d_in[8];
    float* out = (float*)d_out;

    dim3 grid(BATCH * LSEQ);   // 2048 workgroups, one per (b,i)
    dim3 block(128);           // 4 wave32 waves
    hipLaunchKernelGGL(unimol_coord_head_kernel, grid, block, 0, stream,
                       x, coord, nodes, ln_g, ln_b, w1, b1, w2, b2, out);
}
